// RefactoredModel_2619930051278
// MI455X (gfx1250) — matrix operand order repaired and checked by reading, not compiled, not run on hardware
//
#include <hip/hip_runtime.h>
#include <hip/hip_bf16.h>

// ---------------------------------------------------------------------------
// GQA flash-attention for MI455X (gfx1250, wave32, WMMA 16x16x32 bf16->f32).
// B=2, S=2048, H=15, KV=5 (GROUPS=3), D=64. fp32 in/out, bf16 matrix operands,
// f32 accumulation + f32 online softmax. One block (256 thr = 8 waves) owns
// (b, h, 128 q rows); each wave a 16-row q tile; key loop in tiles of 64.
// ---------------------------------------------------------------------------

typedef __attribute__((ext_vector_type(16))) __bf16 v16bf;
typedef __attribute__((ext_vector_type(8)))  __bf16 v8bf;
typedef __attribute__((ext_vector_type(4)))  __bf16 v4bf;
typedef __attribute__((ext_vector_type(8)))  float  v8f;
typedef __attribute__((ext_vector_type(4)))  int    v4i;

#define S_LEN   2048
#define NH      15
#define NKV     5
#define GRP     3
#define HD      64
#define QTILE   128        // q rows per block
#define KTILE   64         // keys per loop iteration (two WMMA K dims)
#define BIG_NEG (-3.40282347e+38f)

// Async global->LDS copy for the mask tile (no dtype conversion needed).
#ifndef USE_ASYNC_MASK
# if defined(__has_builtin)
#  if __has_builtin(__builtin_amdgcn_global_load_async_to_lds_b128)
#   define USE_ASYNC_MASK 1
#  endif
# endif
#endif
#ifndef USE_ASYNC_MASK
# define USE_ASYNC_MASK 0
#endif

#if USE_ASYNC_MASK
// b128 payload pointers: global source (AS1) and LDS destination (AS3).
typedef __attribute__((address_space(1))) v4i GLv4i;
typedef __attribute__((address_space(3))) v4i LDSv4i;
#endif

__launch_bounds__(256)
__global__ void gqa_flash_wmma(const unsigned char* __restrict__ mask,  // [B,S,S] bool8
                               const float*         __restrict__ qg,    // [B,S,NH,HD]
                               const float*         __restrict__ kg,    // [B,S,NKV,HD]
                               const float*         __restrict__ vg,    // [B,S,NKV,HD]
                               float*               __restrict__ out)   // [B,S,NH*HD]
{
    __shared__ __align__(16) __bf16        sK [KTILE * HD];     // [k][d]   row-major
    __shared__ __align__(16) __bf16        sVt[HD * KTILE];     // [d][k]   transposed
    __shared__ __align__(16) unsigned char sM [QTILE * KTILE];  // [q][k]
    __shared__ __align__(16) __bf16        sP [8][16 * KTILE];  // per-wave P staging

    const int t  = threadIdx.x;
    const int w  = t >> 5;          // wave id 0..7
    const int ln = t & 15;          // lane low 4 bits (column / row-in-A)
    const int hi = (t >> 4) & 1;    // half-wave select

    int blk = blockIdx.x;
    const int qb  = blk % (S_LEN / QTILE); blk /= (S_LEN / QTILE);
    const int h   = blk % NH;
    const int b   = blk / NH;
    const int kvh = h / GRP;
    const int q0  = qb * QTILE;

    // ---- Q A-fragments (contiguous-k 16x32 bf16 A layout); 1/sqrt(64) folded in ----
    // lane L: row M = L&15; elems 0..15 -> d = 16*hi + [0,16)   (+32 for aq[1])
    const int   abase = 16 * hi;
    const int   qrowA = q0 + w * 16 + ln;
    const float* qp   = qg + (((size_t)(b * S_LEN + qrowA)) * NH + h) * HD;
    v16bf aq[2];
#pragma unroll
    for (int c = 0; c < 4; ++c) {
        const float* p = qp + (c >> 1) * 32 + (c & 1) * 8 + abase;
#pragma unroll
        for (int i = 0; i < 8; ++i)
            aq[c >> 1][(c & 1) * 8 + i] = (__bf16)(p[i] * 0.125f);
    }

    // ---- flash-attention state ----
    v8f   o[4];                       // O accumulators: 4 d-chunks of 16, C layout
    float m[8], l[8];
#pragma unroll
    for (int c = 0; c < 4; ++c)
#pragma unroll
        for (int e = 0; e < 8; ++e) o[c][e] = 0.0f;
#pragma unroll
    for (int e = 0; e < 8; ++e) { m[e] = BIG_NEG; l[e] = 0.0f; }

    // cooperative-load geometry (256 threads)
    const int krow = t >> 2, kcol = (t & 3) * 16;  // K: 16 f32 per thread
    const int vbr  = (t >> 4) * 4, vbc = (t & 15) * 4; // V: 4x4 block per thread
    const int mrow = t >> 1, mcol = (t & 1) * 32;  // mask: 32 B per thread

    for (int kt = 0; kt < S_LEN; kt += KTILE) {
        __syncthreads();   // previous tile fully consumed before overwrite

        const float* gk = kg + (((size_t)(b * S_LEN + kt + krow)) * NKV + kvh) * HD + kcol;
        const float* gv = vg + (((size_t)(b * S_LEN + kt + vbr))  * NKV + kvh) * HD + vbc;
        const unsigned char* gm = mask + ((size_t)(b * S_LEN + q0 + mrow)) * S_LEN + kt + mcol;

        if (kt + KTILE < S_LEN) {      // -> global_prefetch_b8 of next tile
            __builtin_prefetch(gk + (size_t)KTILE * NKV * HD, 0, 0);
            __builtin_prefetch(gv + (size_t)KTILE * NKV * HD, 0, 0);
            __builtin_prefetch(gm + KTILE, 0, 0);
        }

        // mask tile: async DMA to LDS when available, else VGPR round-trip
#if USE_ASYNC_MASK
        {
            unsigned char* dst = &sM[mrow * KTILE + mcol];
            __builtin_amdgcn_global_load_async_to_lds_b128(
                (GLv4i*)gm, (LDSv4i*)dst, 0, 0);
            __builtin_amdgcn_global_load_async_to_lds_b128(
                (GLv4i*)(gm + 16), (LDSv4i*)(dst + 16), 0, 0);
        }
#else
        {
            uint4* dst = (uint4*)&sM[mrow * KTILE + mcol];
            dst[0] = ((const uint4*)gm)[0];
            dst[1] = ((const uint4*)gm)[1];
        }
#endif

        // K tile -> LDS row-major bf16 (16 f32 -> one 32B LDS store)
        {
            v16bf kb;
#pragma unroll
            for (int i = 0; i < 16; ++i) kb[i] = (__bf16)gk[i];
            *(v16bf*)&sK[krow * HD + kcol] = kb;
        }
        // V tile -> LDS transposed [d][k] bf16, 4x4 block -> 4x ds_store_b64
        {
            float vv[4][4];
#pragma unroll
            for (int i = 0; i < 4; ++i) {
                const float4 f = *(const float4*)(gv + (size_t)i * (NKV * HD));
                vv[i][0] = f.x; vv[i][1] = f.y; vv[i][2] = f.z; vv[i][3] = f.w;
            }
#pragma unroll
            for (int j = 0; j < 4; ++j) {
                v4bf col;
#pragma unroll
                for (int i = 0; i < 4; ++i) col[i] = (__bf16)vv[i][j];
                *(v4bf*)&sVt[(vbc + j) * KTILE + vbr] = col;
            }
        }

#if USE_ASYNC_MASK
        asm volatile("s_wait_asynccnt 0" ::: "memory");
#endif
        __syncthreads();

        // ---- scores: four 16x16 subtiles, each a K=64 chain of 2 WMMAs ----
        const int dbase = 16 * hi;     // B-frag: lanes 16-31 cover upper 16 of K dim
        v8f s[4];
#pragma unroll
        for (int ns = 0; ns < 4; ++ns) {
            const int key = ns * 16 + ln;
            v16bf b0 = *(const v16bf*)&sK[key * HD + dbase];        // d in [0,32)
            v16bf b1 = *(const v16bf*)&sK[key * HD + 32 + dbase];   // d in [32,64)
            v8f acc;
#pragma unroll
            for (int e = 0; e < 8; ++e) acc[e] = 0.0f;
            acc = __builtin_amdgcn_wmma_f32_16x16x32_bf16(false, aq[0], false, b0,
                                                          (short)0, acc, false, false);
            acc = __builtin_amdgcn_wmma_f32_16x16x32_bf16(false, aq[1], false, b1,
                                                          (short)0, acc, false, false);
            s[ns] = acc;
        }

        // ---- mask (C layout: VGPR e -> row e+8*hi, lane&15 -> col) ----
#pragma unroll
        for (int e = 0; e < 8; ++e) {
            const int r = (w * 16 + e + 8 * hi) * KTILE;
#pragma unroll
            for (int ns = 0; ns < 4; ++ns)
                if (!sM[r + ns * 16 + ln]) s[ns][e] = BIG_NEG;
        }

        // ---- online softmax update; P -> per-wave LDS staging ----
        float alpha[8];
#pragma unroll
        for (int e = 0; e < 8; ++e) {
            float rmax = fmaxf(fmaxf(s[0][e], s[1][e]), fmaxf(s[2][e], s[3][e]));
            rmax = fmaxf(rmax, __shfl_xor(rmax, 1, 32));
            rmax = fmaxf(rmax, __shfl_xor(rmax, 2, 32));
            rmax = fmaxf(rmax, __shfl_xor(rmax, 4, 32));
            rmax = fmaxf(rmax, __shfl_xor(rmax, 8, 32));   // stays inside 16-lane half
            const float mn = fmaxf(m[e], rmax);
            const float a  = __expf(m[e] - mn);
            float p[4];
#pragma unroll
            for (int ns = 0; ns < 4; ++ns) p[ns] = __expf(s[ns][e] - mn);
            float rs = (p[0] + p[1]) + (p[2] + p[3]);
            rs += __shfl_xor(rs, 1, 32);
            rs += __shfl_xor(rs, 2, 32);
            rs += __shfl_xor(rs, 4, 32);
            rs += __shfl_xor(rs, 8, 32);
            l[e] = l[e] * a + rs;
            m[e] = mn;
            alpha[e] = a;
            __bf16* pw = &sP[w][(e + 8 * hi) * KTILE];
#pragma unroll
            for (int ns = 0; ns < 4; ++ns) pw[ns * 16 + ln] = (__bf16)p[ns];
        }
#pragma unroll
        for (int c = 0; c < 4; ++c)
#pragma unroll
            for (int e = 0; e < 8; ++e) o[c][e] *= alpha[e];

        // wave-private LDS round-trip: DS ops are in-order per wave; wait anyway
        asm volatile("s_wait_dscnt 0" ::: "memory");

        // ---- P (16x64) as two A-fragments (keys 0..31 / 32..63) ----
        v16bf ap[2];
#pragma unroll
        for (int half = 0; half < 2; ++half) {
            const int o32 = half * 32;
            v8bf lo = *(const v8bf*)&sP[w][ln * KTILE + o32 + abase];
            v8bf hh = *(const v8bf*)&sP[w][ln * KTILE + o32 + 8 + abase];
#pragma unroll
            for (int i = 0; i < 8; ++i) { ap[half][i] = lo[i]; ap[half][8 + i] = hh[i]; }
        }

        // ---- O += P * V  (4 d-chunks x 2 key halves; contiguous Vt B-frags) ----
        const int kb2 = 16 * hi;
#pragma unroll
        for (int c = 0; c < 4; ++c) {
            v16bf bv0 = *(const v16bf*)&sVt[(c * 16 + ln) * KTILE + kb2];
            v16bf bv1 = *(const v16bf*)&sVt[(c * 16 + ln) * KTILE + 32 + kb2];
            o[c] = __builtin_amdgcn_wmma_f32_16x16x32_bf16(false, ap[0], false, bv0,
                                                           (short)0, o[c], false, false);
            o[c] = __builtin_amdgcn_wmma_f32_16x16x32_bf16(false, ap[1], false, bv1,
                                                           (short)0, o[c], false, false);
        }
    }

    // ---- epilogue: out[b, q, h*64 + d] = O / l  (C layout rows) ----
#pragma unroll
    for (int e = 0; e < 8; ++e) {
        const int   qq  = q0 + w * 16 + e + 8 * hi;
        const float inv = 1.0f / l[e];
        float* op = out + ((size_t)(b * S_LEN + qq)) * (NH * HD) + h * HD + ln;
#pragma unroll
        for (int c = 0; c < 4; ++c)
            op[c * 16] = o[c][e] * inv;
    }
}

extern "C" void kernel_launch(void* const* d_in, const int* in_sizes, int n_in,
                              void* d_out, int out_size, void* d_ws, size_t ws_size,
                              hipStream_t stream) {
    (void)in_sizes; (void)n_in; (void)out_size; (void)d_ws; (void)ws_size;
    const unsigned char* mask = (const unsigned char*)d_in[0]; // bool8 [B,S,S]
    const float* q = (const float*)d_in[1];                    // [B,S,15,64]
    const float* k = (const float*)d_in[2];                    // [B,S,5,64]
    const float* v = (const float*)d_in[3];                    // [B,S,5,64]
    float* o = (float*)d_out;                                  // [B,S,960]

    const int B = 2;
    dim3 grid(B * NH * (S_LEN / QTILE));   // 2*15*16 = 480 blocks
    dim3 block(256);                        // 8 waves (wave32)
    gqa_flash_wmma<<<grid, block, 0, stream>>>(mask, q, k, v, o);
}
